// GAT000mp_58480274702496
// MI455X (gfx1250) — matrix operand order repaired and checked by reading, not compiled, not run on hardware
//
#include <hip/hip_runtime.h>
#include <hip/hip_bf16.h>

// ---------------------------------------------------------------------------
// GATv2 4-layer GNN forward for MI455X (gfx1250, wave32, WMMA).
// Dense GEMMs: bf16 WMMA (v_wmma_f32_16x16x32_bf16) with f32 accum; tiles are
// double-buffered in LDS and streamed with GLOBAL_LOAD_ASYNC_TO_LDS_B128,
// overlapped with compute via partial s_wait_asynccnt (in-order completion).
// Edge softmax/scatter phases use L2-resident f32 atomics (per-layer
// activations fit in the 192MB L2).
// ---------------------------------------------------------------------------

typedef __attribute__((ext_vector_type(16))) __bf16 v16bf;
typedef __attribute__((ext_vector_type(8)))  __bf16 bhalf8;
typedef __attribute__((ext_vector_type(4)))  __bf16 bhalf4;
typedef __attribute__((ext_vector_type(8)))  float  v8f;

#define NEG_SLOPE 0.2f
#define NNODES 8000
#define NEDGES 64000
#define NETOT  (NNODES + NEDGES)   // edges + self loops
#define EDGE_DIM 16
#define NGRAPHS 64

// ---------------- small helpers ----------------
__device__ __forceinline__ __bf16 f2bf(float f) {
  union { float f; unsigned u; } x; x.f = f;
  unsigned r = x.u + 0x7FFFu + ((x.u >> 16) & 1u);
  unsigned short s = (unsigned short)(r >> 16);
  return __builtin_bit_cast(__bf16, s);
}
// order-preserving float<->uint map for atomic max on signed floats
__device__ __forceinline__ unsigned fOrd(float f) {
  unsigned u = __float_as_uint(f);
  return (u & 0x80000000u) ? ~u : (u | 0x80000000u);
}
__device__ __forceinline__ float fUnord(unsigned s) {
  unsigned u = (s & 0x80000000u) ? (s & 0x7FFFFFFFu) : ~s;
  return __uint_as_float(u);
}
#define NEG_INF_ORD 0x007FFFFFu   // fOrd(-inf)

// gfx1250 async copy global -> LDS, 16B per lane, tracked by ASYNCcnt.
__device__ __forceinline__ void async_g2l_b128(void* lds_ptr, const void* gptr) {
  unsigned loff = (unsigned)(size_t)lds_ptr;   // low 32 bits of generic ptr = LDS offset
  asm volatile("global_load_async_to_lds_b128 %0, %1, off"
               :: "v"(loff), "v"(gptr) : "memory");
}
// wait until at most `n` async ops outstanding (loads complete in order)
__device__ __forceinline__ void wait_async3() {
  asm volatile("s_wait_asynccnt 0x3" ::: "memory");
}

// ---------------- utility kernels ----------------
__global__ void k_zero_f(float* p, long long n) {
  long long i = (long long)blockIdx.x * blockDim.x + threadIdx.x;
  if (i < n) p[i] = 0.0f;
}
__global__ void k_fill_u(unsigned* p, unsigned v, long long n) {
  long long i = (long long)blockIdx.x * blockDim.x + threadIdx.x;
  if (i < n) p[i] = v;
}
// vectorized f32 -> bf16 (4 elems/thread)
__global__ void k_f2bf4(const float* __restrict__ s, __bf16* __restrict__ d, long long n4) {
  long long i = (long long)blockIdx.x * blockDim.x + threadIdx.x;
  if (i >= n4) return;
  float4 v = ((const float4*)s)[i];
  bhalf4 o = {f2bf(v.x), f2bf(v.y), f2bf(v.z), f2bf(v.w)};
  ((bhalf4*)d)[i] = o;
}
// f32 [K][N] -> bf16 transposed [N][K], LDS-tiled 16x16 (both sides coalesced)
__global__ __launch_bounds__(256) void k_f2bf_T(const float* __restrict__ in,
                                                __bf16* __restrict__ out, int K, int N) {
  __shared__ float t[16][17];
  int tx = threadIdx.x & 15, ty = threadIdx.x >> 4;
  int k0 = blockIdx.y * 16, n0 = blockIdx.x * 16;
  t[ty][tx] = in[(size_t)(k0 + ty) * N + n0 + tx];
  __syncthreads();
  out[(size_t)(n0 + ty) * K + k0 + tx] = f2bf(t[tx][ty]);
}

// ---------------- self-loop edge_attr mean ----------------
__global__ void k_ea_accum(const int* __restrict__ ei, const float* __restrict__ ea,
                           float* __restrict__ sum, float* __restrict__ cnt, int E) {
  int e = blockIdx.x * blockDim.x + threadIdx.x;
  if (e >= E) return;
  int d = ei[E + e];
  const float* row = ea + (size_t)e * EDGE_DIM;
  float* out = sum + (size_t)d * EDGE_DIM;
#pragma unroll
  for (int k = 0; k < EDGE_DIM; ++k) atomicAdd(&out[k], row[k]);
  atomicAdd(&cnt[d], 1.0f);
}
__global__ void k_ea_final(float* __restrict__ sum, const float* __restrict__ cnt, int n16) {
  int i = blockIdx.x * blockDim.x + threadIdx.x;
  if (i >= n16) return;
  float c = cnt[i / EDGE_DIM];
  sum[i] = sum[i] / fmaxf(c, 1.0f);
}

// ---------------- WMMA GEMM: C[M,N] = A[M,K](bf16) * Bt[N,K](bf16)^T + bias --
// Workgroup tile 128x64, 8 waves, each wave owns a 16x64 strip (4 wmma/K-step).
// Double-buffered LDS tiles streamed with global_load_async_to_lds_b128;
// compute on buffer p overlaps the async fill of buffer 1-p.
#define TM 128
#define TN 64
#define TK 32
#define LDA_S 40   // LDS row stride (bf16 elems), 16B-aligned rows, skews banks
#define LDB_S 40

__global__ __launch_bounds__(256) void k_gemm_bf16_wmma(
    const __bf16* __restrict__ A, const __bf16* __restrict__ Bt,
    const float* __restrict__ bias, float* __restrict__ C,
    int M, int N, int K) {
  __shared__ __bf16 lA[2][TM * LDA_S];
  __shared__ __bf16 lB[2][TN * LDB_S];  // lB[.][n][k]

  const int tid  = threadIdx.x;
  const int lane = tid & 31;
  const int wid  = tid >> 5;
  const int mBase = blockIdx.y * TM;
  const int nBase = blockIdx.x * TN;

  // per-thread staging coordinates (3 x 16B async copies per thread per step)
  const int aRow0 = tid >> 2,          aK0 = (tid & 3) * 8;
  const int aRow1 = (tid + 256) >> 2,  aK1 = aK0;          // same (c&3) pattern
  const int bN    = tid & 63,          bK  = (tid >> 6) * 8;
  // (No M guard on A: OOB rows read valid workspace garbage; those accumulator
  //  rows are never stored.)
  auto stage = [&](int buf, int kc) {
    async_g2l_b128(&lA[buf][aRow0 * LDA_S + aK0], A + (size_t)(mBase + aRow0) * K + kc + aK0);
    async_g2l_b128(&lA[buf][aRow1 * LDA_S + aK1], A + (size_t)(mBase + aRow1) * K + kc + aK1);
    async_g2l_b128(&lB[buf][bN * LDB_S + bK],     Bt + (size_t)(nBase + bN) * K + kc + bK);
  };

  v8f acc[4] = {};
  const int nSteps = K / TK;

  stage(0, 0);                                   // prologue: ASYNCcnt = 3
  for (int k = 0; k < nSteps; ++k) {
    const int p = k & 1;
    const int kn = (k + 1 < nSteps) ? (k + 1) : (nSteps - 1);  // clamp: uniform loop
    stage(1 - p, kn * TK);                       // ASYNCcnt = 6
    wait_async3();                               // step k's 3 copies have landed
    __syncthreads();

    // --- fragments + WMMA on buffer p ---
    {
      const int r  = lane & 15;
      // A 16x32 bf16: lanes 0-15 -> K {0..7,16..23}; lanes 16-31 -> K {8..15,24..31}
      const int kbA = (lane < 16) ? 0 : 8;
      v16bf a;
      bhalf8* ap = (bhalf8*)&a;
      const __bf16* ab = &lA[p][(16 * wid + r) * LDA_S + kbA];
      ap[0] = *(const bhalf8*)(ab);
      ap[1] = *(const bhalf8*)(ab + 16);
      // B 32x16 bf16: lanes 0-15 -> K {0..7,16..23} (col N=lane); lanes 16-31 -> K {8..15,24..31}
      const int kbB = (lane < 16) ? 0 : 8;
#pragma unroll
      for (int j = 0; j < 4; ++j) {
        v16bf b;
        bhalf8* bp2 = (bhalf8*)&b;
        const __bf16* bb = &lB[p][(16 * j + r) * LDB_S + kbB];
        bp2[0] = *(const bhalf8*)(bb);
        bp2[1] = *(const bhalf8*)(bb + 16);
        acc[j] = __builtin_amdgcn_wmma_f32_16x16x32_bf16(
            false, a, false, b, (short)0, acc[j], false, false);
      }
    }
    __syncthreads();   // all waves done reading buffer p -> safe to refill next iter
  }

  // --- store: C f32 16x16: VGPR r -> M=r (lanes 0-15) / M=r+8 (lanes 16-31)
  const int colL   = lane & 15;
  const int rowAdd = (lane < 16) ? 0 : 8;
#pragma unroll
  for (int j = 0; j < 4; ++j) {
    int col = nBase + 16 * j + colL;
    float bv = bias[col];
#pragma unroll
    for (int rr = 0; rr < 8; ++rr) {
      int row = mBase + 16 * wid + rr + rowAdd;
      if (row < M) C[(size_t)row * N + col] = acc[j][rr] + bv;
    }
  }
}

// ---------------- edge attention logits (one wave per edge*head) -------------
// logit[e,h] = sum_c leakyrelu(xl[src,h,c]+xr[dst,h,c]+ (ea[e,:]@We[:,h,c])) * att[h,c]
__global__ __launch_bounds__(256) void k_edge_logits(
    const float* __restrict__ xl, const float* __restrict__ xr,
    const float* __restrict__ eattr, const float* __restrict__ ealoop,
    const int* __restrict__ ei, const float* __restrict__ We,
    const float* __restrict__ att, float* __restrict__ logits,
    unsigned* __restrict__ lmaxU, int E, int Etot, int H, int co, int hc) {
  int widx = blockIdx.x * 8 + (threadIdx.x >> 5);
  if (widx >= Etot * H) return;
  int lane = threadIdx.x & 31;
  int e = widx / H, h = widx - e * H;

  int s, d;
  const float* ea;
  if (e < E) { s = ei[e]; d = ei[E + e]; ea = eattr + (size_t)e * EDGE_DIM; }
  else       { s = d = e - E;            ea = ealoop + (size_t)(e - E) * EDGE_DIM; }

  float ear[EDGE_DIM];
#pragma unroll
  for (int k = 0; k < EDGE_DIM; ++k) ear[k] = ea[k];

  const size_t sb = (size_t)s * hc + h * co;
  const size_t db = (size_t)d * hc + h * co;
  const int base = h * co;
  float acc = 0.0f;
  for (int c = lane; c < co; c += 32) {
    int col = base + c;
    float ee = 0.0f;
#pragma unroll
    for (int k = 0; k < EDGE_DIM; ++k) ee += ear[k] * We[(size_t)k * hc + col];
    float v = xl[sb + c] + xr[db + c] + ee;
    v = (v > 0.0f) ? v : NEG_SLOPE * v;
    acc += v * att[col];
  }
#pragma unroll
  for (int o = 16; o > 0; o >>= 1) acc += __shfl_xor(acc, o, 32);
  if (lane == 0) {
    logits[widx] = acc;
    atomicMax(&lmaxU[(size_t)d * H + h], fOrd(acc));
  }
}

// ---------------- softmax numerator + denominator ----------------
__global__ void k_softmax_p(const float* __restrict__ logits,
                            const unsigned* __restrict__ lmaxU,
                            const int* __restrict__ ei,
                            float* __restrict__ pbuf, float* __restrict__ denom,
                            int E, int Etot, int H) {
  int i = blockIdx.x * blockDim.x + threadIdx.x;
  if (i >= Etot * H) return;
  int e = i / H, h = i - e * H;
  int d = (e < E) ? ei[E + e] : (e - E);
  float lm = fUnord(lmaxU[(size_t)d * H + h]);
  float p = __expf(logits[i] - lm);
  pbuf[i] = p;
  atomicAdd(&denom[(size_t)d * H + h], p);
}

// ---------------- weighted scatter: out[dst] += alpha * xl[src] --------------
__global__ __launch_bounds__(256) void k_scatter(
    const float* __restrict__ xl, const float* __restrict__ pbuf,
    const float* __restrict__ denom, const int* __restrict__ ei,
    float* __restrict__ out, int E, int Etot, int H, int co, int hc) {
  int widx = blockIdx.x * 8 + (threadIdx.x >> 5);
  if (widx >= Etot * H) return;
  int lane = threadIdx.x & 31;
  int e = widx / H, h = widx - e * H;
  int s, d;
  if (e < E) { s = ei[e]; d = ei[E + e]; } else { s = d = e - E; }
  float alpha = pbuf[widx] / denom[(size_t)d * H + h];
  const size_t sb = (size_t)s * hc + h * co;
  const size_t db = (size_t)d * hc + h * co;
  for (int c = lane; c < co; c += 32)
    atomicAdd(&out[db + c], alpha * xl[sb + c]);
}

// ---------------- bias + relu (in place) ----------------
__global__ void k_bias_relu(float* __restrict__ h, const float* __restrict__ bias,
                            long long total, int hc) {
  long long i = (long long)blockIdx.x * blockDim.x + threadIdx.x;
  if (i >= total) return;
  h[i] = fmaxf(h[i] + bias[i % hc], 0.0f);
}

// ---------------- global mean pool ----------------
__global__ void k_pool(const float* __restrict__ h, const int* __restrict__ batch,
                       float* __restrict__ pool, float* __restrict__ pcnt,
                       int n, int C) {
  int i = blockIdx.x * blockDim.x + threadIdx.x;
  if (i >= n * C) return;
  int node = i / C, c = i - node * C;
  int b = batch[node];
  atomicAdd(&pool[(size_t)b * C + c], h[i]);
  if (c == 0) atomicAdd(&pcnt[b], 1.0f);
}

// ---------------- final MLP ----------------
__global__ void k_fc1(const float* __restrict__ pool, const float* __restrict__ pcnt,
                      const float* __restrict__ w, const float* __restrict__ b,
                      float* __restrict__ g1) {
  int i = blockIdx.x * blockDim.x + threadIdx.x;
  if (i >= NGRAPHS * 64) return;
  int g = i >> 6, j = i & 63;
  float inv = 1.0f / fmaxf(pcnt[g], 1.0f);
  float s = b[j];
  for (int k = 0; k < 256; ++k) s += pool[(size_t)g * 256 + k] * inv * w[(size_t)k * 64 + j];
  g1[i] = fmaxf(s, 0.0f);
}
__global__ void k_fc2(const float* __restrict__ g1, const float* __restrict__ w,
                      const float* __restrict__ b, float* __restrict__ out) {
  int i = blockIdx.x * blockDim.x + threadIdx.x;
  if (i >= NGRAPHS) return;
  float s = b[0];
  for (int j = 0; j < 64; ++j) s += g1[(size_t)i * 64 + j] * w[j];
  out[i] = s;
}

// ---------------------------------------------------------------------------
// host-side orchestration
// ---------------------------------------------------------------------------
struct LayerCfg { int ci, hc, H, co; int iWl, iBl, iWr, iBr, iWe, iAtt, iBias; };

static inline int cdiv(long long a, long long b) { return (int)((a + b - 1) / b); }

extern "C" void kernel_launch(void* const* d_in, const int* in_sizes, int n_in,
                              void* d_out, int out_size, void* d_ws, size_t ws_size,
                              hipStream_t stream) {
  (void)in_sizes; (void)n_in; (void)out_size; (void)ws_size;

  const float* x       = (const float*)d_in[0];
  const int*   ei      = (const int*)d_in[1];     // [2, E]
  const float* eattr   = (const float*)d_in[2];   // [E, 16]
  const int*   batch   = (const int*)d_in[3];
  const float* fc1_w   = (const float*)d_in[32];
  const float* fc1_b   = (const float*)d_in[33];
  const float* fc2_w   = (const float*)d_in[34];
  const float* fc2_b   = (const float*)d_in[35];
  float*       out     = (float*)d_out;

  const LayerCfg L[4] = {
      {256, 1536, 3, 512,  4, 5, 6, 7, 8, 9, 10},
      {1536, 1024, 2, 512, 11, 12, 13, 14, 15, 16, 17},
      {1024, 512, 2, 256,  18, 19, 20, 21, 22, 23, 24},
      {512, 256, 1, 256,   25, 26, 27, 28, 29, 30, 31},
  };

  // ---- workspace carve-up (256B aligned) ----
  char* wsp = (char*)d_ws;
  size_t cur = 0;
  auto alloc = [&](size_t bytes) -> void* {
    void* p = wsp + cur;
    cur += (bytes + 255) & ~(size_t)255;
    return p;
  };
  const size_t BIG = (size_t)NNODES * 1536;             // max activation elems
  float*  bufA   = (float*)alloc(BIG * 4);
  float*  bufB   = (float*)alloc(BIG * 4);
  float*  xl     = (float*)alloc(BIG * 4);
  float*  xr     = (float*)alloc(BIG * 4);
  __bf16* xbf    = (__bf16*)alloc(BIG * 2);
  __bf16* wlbf   = (__bf16*)alloc((size_t)1536 * 1536 * 2);  // transposed [N][K]
  __bf16* wrbf   = (__bf16*)alloc((size_t)1536 * 1536 * 2);  // transposed [N][K]
  float*  easum  = (float*)alloc((size_t)NNODES * EDGE_DIM * 4);  // becomes ea_loop
  float*  eacnt  = (float*)alloc((size_t)NNODES * 4);
  float*  logits = (float*)alloc((size_t)NETOT * 3 * 4);
  float*  pbuf   = (float*)alloc((size_t)NETOT * 3 * 4);
  unsigned* lmaxU = (unsigned*)alloc((size_t)NNODES * 3 * 4);
  float*  denom  = (float*)alloc((size_t)NNODES * 3 * 4);
  float*  pool   = (float*)alloc((size_t)NGRAPHS * 256 * 4);
  float*  pcnt   = (float*)alloc((size_t)NGRAPHS * 4);
  float*  g1     = (float*)alloc((size_t)NGRAPHS * 64 * 4);

  const int B = 256;

  // ---- self-loop edge_attr mean ----
  k_zero_f<<<cdiv((long long)NNODES * EDGE_DIM, B), B, 0, stream>>>(easum, (long long)NNODES * EDGE_DIM);
  k_zero_f<<<cdiv(NNODES, B), B, 0, stream>>>(eacnt, NNODES);
  k_ea_accum<<<cdiv(NEDGES, B), B, 0, stream>>>(ei, eattr, easum, eacnt, NEDGES);
  k_ea_final<<<cdiv((long long)NNODES * EDGE_DIM, B), B, 0, stream>>>(easum, eacnt, NNODES * EDGE_DIM);

  // ---- layers ----
  const float* hin = x;
  for (int l = 0; l < 4; ++l) {
    const LayerCfg& c = L[l];
    float* hout = (l % 2 == 0) ? bufA : bufB;
    const long long nIn = (long long)NNODES * c.ci;
    const long long nOut = (long long)NNODES * c.hc;
    const long long nEH = (long long)NETOT * c.H;

    // bf16 conversions (weights transposed to [hc][ci] for the GEMM B operand)
    k_f2bf4<<<cdiv(nIn / 4, B), B, 0, stream>>>(hin, xbf, nIn / 4);
    dim3 gt(c.hc / 16, c.ci / 16);
    k_f2bf_T<<<gt, B, 0, stream>>>((const float*)d_in[c.iWl], wlbf, c.ci, c.hc);
    k_f2bf_T<<<gt, B, 0, stream>>>((const float*)d_in[c.iWr], wrbf, c.ci, c.hc);

    // xl = x@Wl + bl ; xr = x@Wr + br    (WMMA)
    dim3 gg(c.hc / TN, cdiv(NNODES, TM));
    k_gemm_bf16_wmma<<<gg, B, 0, stream>>>(xbf, wlbf, (const float*)d_in[c.iBl], xl, NNODES, c.hc, c.ci);
    k_gemm_bf16_wmma<<<gg, B, 0, stream>>>(xbf, wrbf, (const float*)d_in[c.iBr], xr, NNODES, c.hc, c.ci);

    // attention
    k_fill_u<<<cdiv((long long)NNODES * c.H, B), B, 0, stream>>>(lmaxU, NEG_INF_ORD, (long long)NNODES * c.H);
    k_zero_f<<<cdiv((long long)NNODES * c.H, B), B, 0, stream>>>(denom, (long long)NNODES * c.H);
    k_edge_logits<<<cdiv(nEH, 8), B, 0, stream>>>(
        xl, xr, eattr, easum, ei, (const float*)d_in[c.iWe], (const float*)d_in[c.iAtt],
        logits, lmaxU, NEDGES, NETOT, c.H, c.co, c.hc);
    k_softmax_p<<<cdiv(nEH, B), B, 0, stream>>>(logits, lmaxU, ei, pbuf, denom, NEDGES, NETOT, c.H);

    // aggregate
    k_zero_f<<<cdiv(nOut, B), B, 0, stream>>>(hout, nOut);
    k_scatter<<<cdiv(nEH, 8), B, 0, stream>>>(xl, pbuf, denom, ei, hout, NEDGES, NETOT, c.H, c.co, c.hc);
    k_bias_relu<<<cdiv(nOut, B), B, 0, stream>>>(hout, (const float*)d_in[c.iBias], nOut, c.hc);

    hin = hout;
  }

  // ---- mean pool + MLP ----
  k_zero_f<<<cdiv(NGRAPHS * 256, B), B, 0, stream>>>(pool, NGRAPHS * 256);
  k_zero_f<<<1, B, 0, stream>>>(pcnt, NGRAPHS);
  k_pool<<<cdiv((long long)NNODES * 256, B), B, 0, stream>>>(hin, batch, pool, pcnt, NNODES, 256);
  k_fc1<<<cdiv(NGRAPHS * 64, B), B, 0, stream>>>(pool, pcnt, fc1_w, fc1_b, g1);
  k_fc2<<<1, B, 0, stream>>>(g1, fc2_w, fc2_b, out);
}
